// HGCN_30666066494231
// MI455X (gfx1250) — matrix operand order repaired and checked by reading, not compiled, not run on hardware
//
#include <hip/hip_runtime.h>
#include <hip/hip_bf16.h>
#include <math.h>

#define N_NODES 100000
#define N_EDGES 1600000
#define D_IN    1433
#define K1PAD   1440      // 45 * 32
#define NK1     45
#define D1      100
#define D1PAD   112       // 7 * 16
#define S1      128       // padded stride for t1/agg1/h2 (also K for layer-2 GEMM)
#define D2      64
#define NCLS    7
#define MIN_NORM 1e-15f
#define MAXN     (1.0f - 4e-3f)

typedef __bf16 bf16_t;
typedef __bf16 v16bf __attribute__((ext_vector_type(16)));
typedef __bf16 v8bf  __attribute__((ext_vector_type(8)));
typedef float  v8f   __attribute__((ext_vector_type(8)));

typedef unsigned int u32x4 __attribute__((ext_vector_type(4)));
typedef int          i32x8 __attribute__((ext_vector_type(8)));
typedef int          i32x4 __attribute__((ext_vector_type(4)));

#if defined(__has_builtin)
#  if __has_builtin(__builtin_amdgcn_tensor_load_to_lds) && \
      __has_builtin(__builtin_amdgcn_s_wait_tensorcnt)
#    define HAVE_TDM 1
#  endif
#endif
#ifndef HAVE_TDM
#  define HAVE_TDM 0
#endif

__device__ __forceinline__ float artanh_f(float v) {
    v = fminf(fmaxf(v, -1.0f + 1e-7f), 1.0f - 1e-7f);
    return 0.5f * (log1pf(v) - log1pf(-v));
}

#if HAVE_TDM
// Issue a 2-D TDM tile load (global -> LDS). Descriptor packed per CDNA5 ISA §8.3/8.4.
// data_size_code: 0=1B,1=2B,2=4B,3=8B. Dims/strides in data_size units.
// 6-arg builtin form: (g0, g1, g2, g3, g_extra, cpol)
__device__ __forceinline__ void tdm_load_2d(unsigned lds_off, unsigned long long gaddr,
                                            unsigned tensor_d0, unsigned tensor_d1,
                                            unsigned tile_d0, unsigned tile_d1,
                                            unsigned long long stride0,
                                            unsigned data_size_code) {
    u32x4 g0;
    g0[0] = 1u;                                                   // count=1, user desc
    g0[1] = lds_off;                                              // lds_addr (bytes)
    g0[2] = (unsigned)(gaddr & 0xFFFFFFFFu);                      // global_addr[31:0]
    g0[3] = (unsigned)((gaddr >> 32) & 0x1FFFFFFu) | (2u << 30);  // addr[56:32] | type=2
    i32x8 g1;
    g1[0] = (int)(data_size_code << 16);                          // wg_mask=0, data_size
    g1[1] = (int)((tensor_d0 & 0xFFFFu) << 16);                   // tensor_dim0[15:0]
    g1[2] = (int)((tensor_d0 >> 16) | ((tensor_d1 & 0xFFFFu) << 16));
    g1[3] = (int)((tensor_d1 >> 16) | (tile_d0 << 16));           // tile_dim0
    g1[4] = (int)(tile_d1 & 0xFFFFu);                             // tile_dim1, tile_dim2=0
    g1[5] = (int)(stride0 & 0xFFFFFFFFu);                         // tensor_dim0_stride lo
    g1[6] = (int)((stride0 >> 32) & 0xFFFFu);                     // stride hi, dim1_stride=0
    g1[7] = 0;
    i32x4 z4 = {0, 0, 0, 0};
    i32x8 z8 = {0, 0, 0, 0, 0, 0, 0, 0};
    __builtin_amdgcn_tensor_load_to_lds(g0, g1, z4, z4, z8, 0);
}
#endif

// ---------------- row norms of x ----------------
__global__ __launch_bounds__(256)
void k_xnorm(const float* __restrict__ x, float* __restrict__ xnorm) {
    int row = blockIdx.x;
    const float* xr = x + (size_t)row * D_IN;
    float ss = 0.f;
    for (int i = threadIdx.x; i < D_IN; i += 256) { float v = xr[i]; ss += v * v; }
    __shared__ float red[256];
    red[threadIdx.x] = ss;
    __syncthreads();
    for (int s = 128; s > 0; s >>= 1) {
        if (threadIdx.x < s) red[threadIdx.x] += red[threadIdx.x + s];
        __syncthreads();
    }
    if (threadIdx.x == 0) xnorm[row] = sqrtf(red[0]);
}

// ---------------- weight conversion (f32 -> padded bf16) ----------------
__global__ void k_prep_w1(const float* __restrict__ w1, bf16_t* __restrict__ w1b) {
    int total = D1PAD * K1PAD;
    for (int idx = blockIdx.x * blockDim.x + threadIdx.x; idx < total;
         idx += gridDim.x * blockDim.x) {
        int n = idx / K1PAD, k = idx - n * K1PAD;
        float v = (n < D1 && k < D_IN) ? w1[n * D_IN + k] : 0.f;
        w1b[idx] = (bf16_t)v;
    }
}

__global__ void k_prep_w2(const float* __restrict__ w2, bf16_t* __restrict__ w2b) {
    int total = D2 * S1;
    for (int idx = blockIdx.x * blockDim.x + threadIdx.x; idx < total;
         idx += gridDim.x * blockDim.x) {
        int n = idx / S1, k = idx - n * S1;
        float v = (k < D1) ? w2[n * D1 + k] : 0.f;
        w2b[idx] = (bf16_t)v;
    }
}

// ---------------- hyperbolic biases: hb = proj(expmap0(b)) ----------------
__global__ __launch_bounds__(128)
void k_prep_bias(const float* __restrict__ b1, const float* __restrict__ b2,
                 float* __restrict__ hb1, float* __restrict__ hb2) {
    __shared__ float red[128];
    int tid = threadIdx.x;
    float ss = 0.f;
    for (int i = tid; i < D1; i += 128) { float v = b1[i]; ss += v * v; }
    red[tid] = ss; __syncthreads();
    for (int s = 64; s > 0; s >>= 1) { if (tid < s) red[tid] += red[tid + s]; __syncthreads(); }
    float bn = fmaxf(sqrtf(red[0]), MIN_NORM);
    float hn = fminf(tanhf(bn), MAXN);
    float sc = hn / bn;
    for (int i = tid; i < D1PAD; i += 128) hb1[i] = (i < D1) ? b1[i] * sc : 0.f;
    if (tid == 0) hb1[112] = hn * hn;
    __syncthreads();
    float ss2 = 0.f;
    for (int i = tid; i < D2; i += 128) { float v = b2[i]; ss2 += v * v; }
    red[tid] = ss2; __syncthreads();
    for (int s = 64; s > 0; s >>= 1) { if (tid < s) red[tid] += red[tid + s]; __syncthreads(); }
    float bn2 = fmaxf(sqrtf(red[0]), MIN_NORM);
    float hn2 = fminf(tanhf(bn2), MAXN);
    float sc2 = hn2 / bn2;
    if (tid < D2) hb2[tid] = b2[tid] * sc2;
    if (tid == 0) hb2[64] = hn2 * hn2;
}

// ---------------- fused encode + HypLinear1 + logmap0 (WMMA bf16 + TDM B-staging) ----------------
__global__ __launch_bounds__(128)
void k_gemm1(const float* __restrict__ x, const bf16_t* __restrict__ w1b,
             const float* __restrict__ xnorm, const float* __restrict__ hb1,
             float* __restrict__ t1) {
    __shared__ float tile[4][16][116];
    __shared__ float s_hb[D1PAD];
    __shared__ float s_hbsq;
    __shared__ float s_ca[64], s_cb[64];
#if HAVE_TDM
    __shared__ bf16_t lds_b[2][D1PAD * 32];   // double-buffered 112x32 bf16 weight chunk
#endif

    const int tid = threadIdx.x;
    const int wave = tid >> 5, lane = tid & 31;
    const int hi = lane >> 4, lanelo = lane & 15;
    const int mbase = blockIdx.x * 64;
    const int mrow = mbase + wave * 16 + lanelo;
    const int rowc = min(mrow, N_NODES - 1);
    const float* xr = x + (size_t)rowc * D_IN;

    if (tid < D1PAD) s_hb[tid] = hb1[tid];
    if (tid == 0) s_hbsq = hb1[112];

    v8f acc[7];
#pragma unroll
    for (int nt = 0; nt < 7; ++nt) acc[nt] = (v8f){0.f,0.f,0.f,0.f,0.f,0.f,0.f,0.f};

#if HAVE_TDM
    const unsigned long long wadr = (unsigned long long)(uintptr_t)w1b;
    if (wave == 0) {
        // prime buffer 0 with K-chunk 0
        tdm_load_2d((unsigned)(uintptr_t)&lds_b[0][0], wadr,
                    K1PAD, D1PAD, 32, D1PAD, K1PAD, 1);
    }
#endif

    for (int kb = 0; kb < NK1; ++kb) {
        const int kbase = kb * 32;
#if HAVE_TDM
        if (wave == 0) __builtin_amdgcn_s_wait_tensorcnt(0);   // chunk kb landed
        __syncthreads();
        if (wave == 0 && kb + 1 < NK1) {
            // prefetch next chunk into the other buffer while this one is consumed
            tdm_load_2d((unsigned)(uintptr_t)&lds_b[(kb + 1) & 1][0],
                        wadr + (unsigned long long)(kb + 1) * 64ull,
                        K1PAD - (kb + 1) * 32, D1PAD, 32, D1PAD, K1PAD, 1);
        }
#endif
        if (kbase + 32 < D_IN) __builtin_prefetch(xr + kbase + 32, 0, 1);
        // A fragment: 16x32 bf16, lane holds M=lanelo; K = kbase + hi*16 + [0..15]
        v16bf a;
#pragma unroll
        for (int e = 0; e < 8; ++e) {
            int k0 = kbase + hi * 16 + e;
            int k1 = kbase + hi * 16 + 8 + e;
            a[e]     = (bf16_t)((k0 < D_IN) ? xr[k0] : 0.f);
            a[8 + e] = (bf16_t)((k1 < D_IN) ? xr[k1] : 0.f);
        }
#pragma unroll
        for (int nt = 0; nt < 7; ++nt) {
            // B fragment: lane holds N = nt*16+lanelo, K = kbase + hi*16 + [0..15]
#if HAVE_TDM
            const bf16_t* bp = &lds_b[kb & 1][(nt * 16 + lanelo) * 32 + hi * 16];
#else
            const bf16_t* bp = w1b + (size_t)(nt * 16 + lanelo) * K1PAD + kbase + hi * 16;
#endif
            v8bf blo = *(const v8bf*)bp;
            v8bf bhi = *(const v8bf*)(bp + 8);
            v16bf b;
#pragma unroll
            for (int e = 0; e < 8; ++e) { b[e] = blo[e]; b[8 + e] = bhi[e]; }
            acc[nt] = __builtin_amdgcn_wmma_f32_16x16x32_bf16(
                false, a, false, b, (short)0, acc[nt], false, false);
        }
#if HAVE_TDM
        __syncthreads();   // buffer (kb&1) free for reuse at kb+2
#endif
    }

    // spill C tiles: VGPR r -> row hi*8+r, col nt*16+lanelo
#pragma unroll
    for (int nt = 0; nt < 7; ++nt)
#pragma unroll
        for (int r = 0; r < 8; ++r)
            tile[wave][hi * 8 + r][nt * 16 + lanelo] = acc[nt][r];
    __syncthreads();

    if (tid < 64) {
        const float* g = tile[tid >> 4][tid & 15];
        float gss = 0.f, gdot = 0.f;
        for (int n = 0; n < D1PAD; ++n) { float gv = g[n]; gss += gv * gv; gdot += gv * s_hb[n]; }
        int row = mbase + tid;
        float xn   = fmaxf(xnorm[min(row, N_NODES - 1)], MIN_NORM);
        float h0n  = fminf(tanhf(xn), MAXN);          // ||proj(expmap0(x))||
        float s0   = h0n / xn;                        // h0 = s0 * x
        float xpr  = fmaxf(h0n, MIN_NORM);
        float mxn  = s0 * sqrtf(gss);                 // ||mx||
        float mxnc = fmaxf(mxn, MIN_NORM);
        float tt   = tanhf(mxn / xpr * artanh_f(xpr));
        float mvn  = fminf(tt, MAXN);                 // ||proj(res)||
        float beta = s0 * mvn / mxnc;                 // mv = beta * g
        float x2 = mvn * mvn, y2 = s_hbsq, xy = beta * gdot;
        float A_ = 1.f + 2.f * xy + y2;
        float B_ = 1.f - x2;
        float den = fmaxf(1.f + 2.f * xy + x2 * y2, MIN_NORM);
        float ca = A_ * beta / den, cb = B_ / den;    // out = ca*g + cb*hb
        float on2 = sqrtf(fmaxf(ca * ca * gss + 2.f * ca * cb * gdot + cb * cb * y2, 0.f));
        float f1 = fminf(1.f, MAXN / fmaxf(on2, MIN_NORM));
        float h1n = fminf(on2, MAXN);
        float l = artanh_f(h1n) / fmaxf(h1n, MIN_NORM);  // logmap0 scale
        s_ca[tid] = ca * f1 * l;
        s_cb[tid] = cb * f1 * l;
    }
    __syncthreads();

    for (int idx = tid; idx < 64 * S1; idx += 128) {
        int r = idx >> 7, n = idx & (S1 - 1);
        int row = mbase + r;
        if (row < N_NODES) {
            float v = 0.f;
            if (n < D1PAD) v = s_ca[r] * tile[r >> 4][r & 15][n] + s_cb[r] * s_hb[n];
            t1[(size_t)row * S1 + n] = v;
        }
    }
}

// ---------------- zero-fill ----------------
__global__ void k_zero(float4* __restrict__ p, long n4) {
    long i = (long)blockIdx.x * blockDim.x + threadIdx.x;
    if (i < n4) p[i] = make_float4(0.f, 0.f, 0.f, 0.f);
}

// ---------------- edge aggregation: agg[row] += w * t[col] ----------------
__global__ __launch_bounds__(256)
void k_agg(const int* __restrict__ erow, const int* __restrict__ ecol,
           const float* __restrict__ ew, const float* __restrict__ t,
           float* __restrict__ agg, int ncols, int stride) {
    int wid = (int)(((long)blockIdx.x * blockDim.x + threadIdx.x) >> 5);
    int lane = threadIdx.x & 31;
    if (wid >= N_EDGES) return;
    int r = erow[wid], c = ecol[wid];
    float w = ew[wid];
    int base = lane * 4;
    if (base < ncols) {
        float4 v = *(const float4*)(t + (size_t)c * stride + base);
        float* dst = agg + (size_t)r * stride + base;
        atomicAdd(dst + 0, w * v.x);
        atomicAdd(dst + 1, w * v.y);
        atomicAdd(dst + 2, w * v.z);
        atomicAdd(dst + 3, w * v.w);
    }
}

// ---------------- post-agg: expmap0+proj, then HypAct -> hyperbolic h2 + norm ----------------
__global__ __launch_bounds__(256)
void k_act1(const float* __restrict__ agg, float* __restrict__ h2,
            float* __restrict__ h2norm) {
    int wave = threadIdx.x >> 5, lane = threadIdx.x & 31;
    int row = blockIdx.x * 8 + wave;
    const float* ar = agg + (size_t)row * S1;
    float4 v = *(const float4*)(ar + lane * 4);
    float ss  = v.x * v.x + v.y * v.y + v.z * v.z + v.w * v.w;
    float4 p = make_float4(fmaxf(v.x, 0.f), fmaxf(v.y, 0.f), fmaxf(v.z, 0.f), fmaxf(v.w, 0.f));
    float ssp = p.x * p.x + p.y * p.y + p.z * p.z + p.w * p.w;
    for (int o = 16; o > 0; o >>= 1) {
        ss  += __shfl_xor(ss, o, 32);
        ssp += __shfl_xor(ssp, o, 32);
    }
    float rn = fmaxf(sqrtf(ss), MIN_NORM);
    float hn = fminf(tanhf(rn), MAXN);
    float s  = hn / rn;
    float l  = artanh_f(hn) / fmaxf(hn, MIN_NORM);
    float ls = l * s;
    float un = fmaxf(ls * sqrtf(ssp), MIN_NORM);
    float h2n = fminf(tanhf(un), MAXN);
    float s2 = h2n / un;
    float cf = s2 * ls;
    float4 o4 = make_float4(cf * p.x, cf * p.y, cf * p.z, cf * p.w);
    *(float4*)(h2 + (size_t)row * S1 + lane * 4) = o4;
    if (lane == 0) h2norm[row] = (ssp > 0.f) ? h2n : 0.f;
}

// ---------------- fused HypLinear2 + logmap0 (WMMA bf16, K=128, N=64, TDM W-preload) ----------------
__global__ __launch_bounds__(128)
void k_gemm2(const float* __restrict__ h2, const bf16_t* __restrict__ w2b,
             const float* __restrict__ h2norm, const float* __restrict__ hb2,
             float* __restrict__ t2) {
    __shared__ float tile[4][16][68];
    __shared__ float s_hb[D2];
    __shared__ float s_hbsq;
    __shared__ float s_ca[64], s_cb[64];
#if HAVE_TDM
    __shared__ bf16_t lds_w[D2 * S1];   // whole 64x128 bf16 weight (16 KB)
#endif

    const int tid = threadIdx.x;
    const int wave = tid >> 5, lane = tid & 31;
    const int hi = lane >> 4, lanelo = lane & 15;
    const int mbase = blockIdx.x * 64;
    const int mrow = mbase + wave * 16 + lanelo;
    const int rowc = min(mrow, N_NODES - 1);
    const float* hr = h2 + (size_t)rowc * S1;

    if (tid < D2) s_hb[tid] = hb2[tid];
    if (tid == 0) s_hbsq = hb2[64];

#if HAVE_TDM
    if (wave == 0) {
        tdm_load_2d((unsigned)(uintptr_t)&lds_w[0],
                    (unsigned long long)(uintptr_t)w2b,
                    S1, D2, S1, D2, S1, 1);
        __builtin_amdgcn_s_wait_tensorcnt(0);
    }
    __syncthreads();
#endif

    v8f acc[4];
#pragma unroll
    for (int nt = 0; nt < 4; ++nt) acc[nt] = (v8f){0.f,0.f,0.f,0.f,0.f,0.f,0.f,0.f};

#pragma unroll
    for (int kb = 0; kb < 4; ++kb) {
        const int kbase = kb * 32;
        const float4* p0 = (const float4*)(hr + kbase + hi * 16);
        float4 f0 = p0[0], f1 = p0[1];
        const float4* p1 = (const float4*)(hr + kbase + hi * 16 + 8);
        float4 f2 = p1[0], f3 = p1[1];
        v16bf a;
        a[0] = (bf16_t)f0.x;  a[1] = (bf16_t)f0.y;  a[2]  = (bf16_t)f0.z;  a[3]  = (bf16_t)f0.w;
        a[4] = (bf16_t)f1.x;  a[5] = (bf16_t)f1.y;  a[6]  = (bf16_t)f1.z;  a[7]  = (bf16_t)f1.w;
        a[8] = (bf16_t)f2.x;  a[9] = (bf16_t)f2.y;  a[10] = (bf16_t)f2.z;  a[11] = (bf16_t)f2.w;
        a[12] = (bf16_t)f3.x; a[13] = (bf16_t)f3.y; a[14] = (bf16_t)f3.z;  a[15] = (bf16_t)f3.w;
#pragma unroll
        for (int nt = 0; nt < 4; ++nt) {
#if HAVE_TDM
            const bf16_t* bp = &lds_w[(size_t)(nt * 16 + lanelo) * S1 + kbase + hi * 16];
#else
            const bf16_t* bp = w2b + (size_t)(nt * 16 + lanelo) * S1 + kbase + hi * 16;
#endif
            v8bf blo = *(const v8bf*)bp;
            v8bf bhi = *(const v8bf*)(bp + 8);
            v16bf b;
#pragma unroll
            for (int e = 0; e < 8; ++e) { b[e] = blo[e]; b[8 + e] = bhi[e]; }
            acc[nt] = __builtin_amdgcn_wmma_f32_16x16x32_bf16(
                false, a, false, b, (short)0, acc[nt], false, false);
        }
    }

#pragma unroll
    for (int nt = 0; nt < 4; ++nt)
#pragma unroll
        for (int r = 0; r < 8; ++r)
            tile[wave][hi * 8 + r][nt * 16 + lanelo] = acc[nt][r];
    __syncthreads();

    if (tid < 64) {
        const float* g = tile[tid >> 4][tid & 15];
        float gss = 0.f, gdot = 0.f;
        for (int n = 0; n < D2; ++n) { float gv = g[n]; gss += gv * gv; gdot += gv * s_hb[n]; }
        int row = mbase + tid;
        float xn   = fmaxf(h2norm[min(row, N_NODES - 1)], MIN_NORM);
        float mxn  = sqrtf(gss);
        float mxnc = fmaxf(mxn, MIN_NORM);
        float tt   = tanhf(mxn / xn * artanh_f(xn));
        float mvn  = fminf(tt, MAXN);
        float beta = mvn / mxnc;
        float x2 = mvn * mvn, y2 = s_hbsq, xy = beta * gdot;
        float A_ = 1.f + 2.f * xy + y2;
        float B_ = 1.f - x2;
        float den = fmaxf(1.f + 2.f * xy + x2 * y2, MIN_NORM);
        float ca = A_ * beta / den, cb = B_ / den;
        float on2 = sqrtf(fmaxf(ca * ca * gss + 2.f * ca * cb * gdot + cb * cb * y2, 0.f));
        float f1 = fminf(1.f, MAXN / fmaxf(on2, MIN_NORM));
        float h1n = fminf(on2, MAXN);
        float l = artanh_f(h1n) / fmaxf(h1n, MIN_NORM);
        s_ca[tid] = ca * f1 * l;
        s_cb[tid] = cb * f1 * l;
    }
    __syncthreads();

    for (int idx = tid; idx < 64 * D2; idx += 128) {
        int r = idx >> 6, n = idx & 63;
        int row = mbase + r;
        if (row < N_NODES)
            t2[(size_t)row * D2 + n] = s_ca[r] * tile[r >> 4][r & 15][n] + s_cb[r] * s_hb[n];
    }
}

// ---------------- decode: expmap0+proj, logmap0, linear+relu, log_softmax ----------------
__global__ __launch_bounds__(256)
void k_decode(const float* __restrict__ agg2, const float* __restrict__ lin_w,
              const float* __restrict__ lin_b, float* __restrict__ out) {
    __shared__ float s_lw[NCLS * D2];
    __shared__ float s_lb[NCLS];
    for (int i = threadIdx.x; i < NCLS * D2; i += 256) s_lw[i] = lin_w[i];
    if (threadIdx.x < NCLS) s_lb[threadIdx.x] = lin_b[threadIdx.x];
    __syncthreads();

    int wave = threadIdx.x >> 5, lane = threadIdx.x & 31;
    int row = blockIdx.x * 8 + wave;
    const float* ar = agg2 + (size_t)row * D2;
    float2 v = *(const float2*)(ar + lane * 2);
    float ss = v.x * v.x + v.y * v.y;
    for (int o = 16; o > 0; o >>= 1) ss += __shfl_xor(ss, o, 32);
    float rn = fmaxf(sqrtf(ss), MIN_NORM);
    float hn = fminf(tanhf(rn), MAXN);
    float s  = hn / rn;
    float l  = artanh_f(hn) / fmaxf(hn, MIN_NORM);
    float ls = l * s;
    float tx = ls * v.x, ty = ls * v.y;

    float p[NCLS];
#pragma unroll
    for (int j = 0; j < NCLS; ++j) {
        p[j] = tx * s_lw[j * D2 + lane * 2] + ty * s_lw[j * D2 + lane * 2 + 1];
        for (int o = 16; o > 0; o >>= 1) p[j] += __shfl_xor(p[j], o, 32);
    }
    if (lane == 0) {
        float lg[NCLS];
        float m = -1e30f;
#pragma unroll
        for (int j = 0; j < NCLS; ++j) { lg[j] = fmaxf(p[j] + s_lb[j], 0.f); m = fmaxf(m, lg[j]); }
        float sum = 0.f;
#pragma unroll
        for (int j = 0; j < NCLS; ++j) sum += expf(lg[j] - m);
        float lse = m + logf(sum);
#pragma unroll
        for (int j = 0; j < NCLS; ++j) out[(size_t)row * NCLS + j] = lg[j] - lse;
    }
}

extern "C" void kernel_launch(void* const* d_in, const int* in_sizes, int n_in,
                              void* d_out, int out_size, void* d_ws, size_t ws_size,
                              hipStream_t stream) {
    const float* x    = (const float*)d_in[0];
    const int*   erow = (const int*)d_in[1];
    const int*   ecol = (const int*)d_in[2];
    const float* ew   = (const float*)d_in[3];
    const float* w1   = (const float*)d_in[4];
    const float* b1   = (const float*)d_in[5];
    const float* w2   = (const float*)d_in[6];
    const float* b2   = (const float*)d_in[7];
    const float* lw   = (const float*)d_in[8];
    const float* lb   = (const float*)d_in[9];
    float* out = (float*)d_out;

    // workspace carve-up (R0/R1 ping-pong: R0 = t1 -> h2 -> agg2 ; R1 = agg1 -> t2)
    float* R0   = (float*)d_ws;                       // N*S1 f32
    float* R1   = R0 + (size_t)N_NODES * S1;          // N*S1 f32
    float* xn   = R1 + (size_t)N_NODES * S1;          // N f32
    float* h2n  = xn + N_NODES;                       // N f32
    float* hb1  = h2n + N_NODES;                      // 128 f32 (112 vals + ||hb||^2)
    float* hb2  = hb1 + 128;                          // 80 f32 (64 vals + ||hb||^2)
    bf16_t* w1b = (bf16_t*)(hb2 + 80);                // 112*1440 bf16
    bf16_t* w2b = w1b + (size_t)D1PAD * K1PAD;        // 64*128 bf16

    // prep
    k_xnorm<<<N_NODES, 256, 0, stream>>>(x, xn);
    k_prep_w1<<<640, 256, 0, stream>>>(w1, w1b);
    k_prep_w2<<<32, 256, 0, stream>>>(w2, w2b);
    k_prep_bias<<<1, 128, 0, stream>>>(b1, b2, hb1, hb2);

    // layer 1
    k_gemm1<<<(N_NODES + 63) / 64, 128, 0, stream>>>(x, w1b, xn, hb1, /*t1*/ R0);
    {
        long n4 = (long)N_NODES * S1 / 4;
        k_zero<<<(int)((n4 + 255) / 256), 256, 0, stream>>>((float4*)R1, n4);
    }
    k_agg<<<N_EDGES / 8, 256, 0, stream>>>(erow, ecol, ew, /*t1*/ R0, /*agg1*/ R1, D1PAD, S1);
    k_act1<<<N_NODES / 8, 256, 0, stream>>>(/*agg1*/ R1, /*h2*/ R0, h2n);

    // layer 2
    k_gemm2<<<(N_NODES + 63) / 64, 128, 0, stream>>>(/*h2*/ R0, w2b, h2n, hb2, /*t2*/ R1);
    {
        long n4 = (long)N_NODES * D2 / 4;
        k_zero<<<(int)((n4 + 255) / 256), 256, 0, stream>>>((float4*)R0, n4);
    }
    k_agg<<<N_EDGES / 8, 256, 0, stream>>>(erow, ecol, ew, /*t2*/ R1, /*agg2*/ R0, D2, D2);

    // decode
    k_decode<<<N_NODES / 8, 256, 0, stream>>>(/*agg2*/ R0, lw, lb, out);
}
